// SimplicialConvolutionalNetwork_82231443849936
// MI455X (gfx1250) — matrix operand order repaired and checked by reading, not compiled, not run on hardware
//
#include <hip/hip_runtime.h>
#include <math.h>

// ---------------------------------------------------------------------------
// CDNA5 WMMA types (wave32): v_wmma_f32_16x16x32_bf16
// ---------------------------------------------------------------------------
typedef __attribute__((ext_vector_type(16))) __bf16 v16bf;
typedef __attribute__((ext_vector_type(8)))  __bf16 v8bf;
typedef __attribute__((ext_vector_type(8)))  float  v8f;

__device__ __forceinline__ __bf16 f2bf(float f) {
    unsigned u = __builtin_bit_cast(unsigned, f);
    unsigned r = u + 0x7FFFu + ((u >> 16) & 1u);   // round-to-nearest-even
    return __builtin_bit_cast(__bf16, (unsigned short)(r >> 16));
}
__device__ __forceinline__ float bf2f(__bf16 b) {
    unsigned u = ((unsigned)__builtin_bit_cast(unsigned short, b)) << 16;
    return __builtin_bit_cast(float, u);
}

// ---------------------------------------------------------------------------
// Elementwise f32 -> bf16 (8 elems/thread; n must be a multiple of 8 — all
// our arrays are multiples of 128).
// ---------------------------------------------------------------------------
__global__ __launch_bounds__(256) void cvt_bf16_k(
    const float* __restrict__ in, __bf16* __restrict__ out, long n)
{
    long i = ((long)blockIdx.x * 256 + threadIdx.x) * 8;
    if (i >= n) return;
    float4 a = *(const float4*)(in + i);
    float4 b = *(const float4*)(in + i + 4);
    v8bf o;
    o[0] = f2bf(a.x); o[1] = f2bf(a.y); o[2] = f2bf(a.z); o[3] = f2bf(a.w);
    o[4] = f2bf(b.x); o[5] = f2bf(b.y); o[6] = f2bf(b.z); o[7] = f2bf(b.w);
    *(v8bf*)(out + i) = o;
}

// ---------------------------------------------------------------------------
// Dense GEMM: Y[M,128](f32) = Xb[M,128](bf16) @ Wb[128,128](bf16)
// 256 threads = 8 wave32; wave owns a 16-row strip of a 128x128 tile.
// A frags: two 16B global_load_b128 per K-chunk (no conversion VALU).
// W staged in LDS transposed [n][k], pitch 136 (16B rows, bank-shifted).
// ---------------------------------------------------------------------------
#define LDS_PITCH 136

__global__ __launch_bounds__(256) void gemm128_wmma(
    const __bf16* __restrict__ Xb, const __bf16* __restrict__ Wb,
    float* __restrict__ Y, int M)
{
    __shared__ __bf16 Wl[128 * LDS_PITCH];

    const int tid = threadIdx.x;
    // Stage W (row-major [k][n] bf16) into LDS as [n][k]; coalesced reads.
    for (int i = tid; i < 128 * 128; i += 256) {
        int k = i >> 7, n = i & 127;
        Wl[n * LDS_PITCH + k] = Wb[i];
    }
    __syncthreads();

    const int wave = tid >> 5;
    const int lane = tid & 31;
    const int half = lane >> 4;
    const int l16  = lane & 15;
    const int tileRow = blockIdx.x * 128 + wave * 16;

    v8f acc[8];
#pragma unroll
    for (int j = 0; j < 8; ++j)
#pragma unroll
        for (int i = 0; i < 8; ++i) acc[j][i] = 0.0f;

    int arow = tileRow + l16;
    arow = (arow < M) ? arow : (M - 1);            // uniform clamp, EXEC stays full
    const __bf16* __restrict__ xr = Xb + (size_t)arow * 128;

#pragma unroll
    for (int kb = 0; kb < 4; ++kb) {
        // A frag (ISA 16-bit A 16x32): elems 0..7 -> K=kb*32+8h+e,
        //                              elems 8..15 -> K=kb*32+16+8h+(e-8)
        const int kbase = kb * 32 + half * 8;
        v8bf alo = *(const v8bf*)(xr + kbase);
        v8bf ahi = *(const v8bf*)(xr + kbase + 16);
        v16bf a = __builtin_shufflevector(alo, ahi,
            0, 1, 2, 3, 4, 5, 6, 7, 8, 9, 10, 11, 12, 13, 14, 15);

        // B frag: elems 0..7 -> K = kb*32 + 8*half + e, elems 8..15 -> K = kb*32 + 16 + 8*half + (e-8), N = j*16 + l16
        const int kcol = kb * 32 + half * 8;
#pragma unroll
        for (int j = 0; j < 8; ++j) {
            const __bf16* bp = &Wl[(j * 16 + l16) * LDS_PITCH + kcol];
            v8bf blo = *(const v8bf*)bp;
            v8bf bhi = *(const v8bf*)(bp + 16);
            v16bf b = __builtin_shufflevector(blo, bhi,
                0, 1, 2, 3, 4, 5, 6, 7, 8, 9, 10, 11, 12, 13, 14, 15);
            acc[j] = __builtin_amdgcn_wmma_f32_16x16x32_bf16(
                false, a, false, b, (short)0, acc[j], false, false);
        }
    }

    // D layout: VGPR i -> row = tileRow + i + 8*half, col = j*16 + l16
#pragma unroll
    for (int j = 0; j < 8; ++j) {
#pragma unroll
        for (int i = 0; i < 8; ++i) {
            int r = tileRow + i + 8 * half;
            if (r < M) Y[(size_t)r * 128 + j * 16 + l16] = acc[j][i];
        }
    }
}

// ---------------------------------------------------------------------------
// COO SpMM scatter: Y[rows[e], :] += vals[e] * T[cols[e], :]   (128 channels)
// One wave per nnz; lane owns 4 channels. Feature state is L2-resident.
// ---------------------------------------------------------------------------
__global__ __launch_bounds__(256) void spmm_atomic(
    const int* __restrict__ rows, const int* __restrict__ cols,
    const float* __restrict__ vals, const float* __restrict__ T,
    float* __restrict__ Y, int nnz)
{
    int t = blockIdx.x * 256 + threadIdx.x;
    int e = t >> 5;
    int lane = t & 31;
    if (e >= nnz) return;
    int r = rows[e];
    int c = cols[e];
    float v = vals[e];
    const float4 s = ((const float4*)(T + (size_t)c * 128))[lane];
    float* dst = Y + (size_t)r * 128 + lane * 4;
    atomicAdd(dst + 0, v * s.x);
    atomicAdd(dst + 1, v * s.y);
    atomicAdd(dst + 2, v * s.z);
    atomicAdd(dst + 3, v * s.w);
}

// ---------------------------------------------------------------------------
// Fused sigmoid + bf16 downconvert: out_bf16 = sigmoid(acc_f32)
// ---------------------------------------------------------------------------
__global__ __launch_bounds__(256) void sig_bf16_k(
    const float* __restrict__ acc, __bf16* __restrict__ out, long n)
{
    long i = ((long)blockIdx.x * 256 + threadIdx.x) * 8;
    if (i >= n) return;
    float4 a = *(const float4*)(acc + i);
    float4 b = *(const float4*)(acc + i + 4);
    v8bf o;
    o[0] = f2bf(1.0f / (1.0f + __expf(-a.x)));
    o[1] = f2bf(1.0f / (1.0f + __expf(-a.y)));
    o[2] = f2bf(1.0f / (1.0f + __expf(-a.z)));
    o[3] = f2bf(1.0f / (1.0f + __expf(-a.w)));
    o[4] = f2bf(1.0f / (1.0f + __expf(-b.x)));
    o[5] = f2bf(1.0f / (1.0f + __expf(-b.y)));
    o[6] = f2bf(1.0f / (1.0f + __expf(-b.z)));
    o[7] = f2bf(1.0f / (1.0f + __expf(-b.w)));
    *(v8bf*)(out + i) = o;
}

// ---------------------------------------------------------------------------
// Head: out = softmax(Xb @ W_out + b_out), Xb:[M,128] bf16, W_out:[128,10] f32
// ---------------------------------------------------------------------------
__global__ __launch_bounds__(256) void head_k(
    const __bf16* __restrict__ Xb, const float* __restrict__ Wo,
    const float* __restrict__ bo, float* __restrict__ out, int M)
{
    int r = blockIdx.x * 256 + threadIdx.x;
    if (r >= M) return;
    float lg[10];
#pragma unroll
    for (int j = 0; j < 10; ++j) lg[j] = bo[j];
    const __bf16* __restrict__ x = Xb + (size_t)r * 128;
    for (int c = 0; c < 128; ++c) {
        float xv = bf2f(x[c]);
#pragma unroll
        for (int j = 0; j < 10; ++j) lg[j] += xv * Wo[c * 10 + j];
    }
    float mx = lg[0];
#pragma unroll
    for (int j = 1; j < 10; ++j) mx = fmaxf(mx, lg[j]);
    float sum = 0.0f;
#pragma unroll
    for (int j = 0; j < 10; ++j) { lg[j] = __expf(lg[j] - mx); sum += lg[j]; }
    float inv = 1.0f / sum;
#pragma unroll
    for (int j = 0; j < 10; ++j) out[(size_t)r * 10 + j] = lg[j] * inv;
}

// ---------------------------------------------------------------------------
// Orchestration
// ---------------------------------------------------------------------------
static inline void launch_cvt(const float* in, __bf16* out, long n, hipStream_t s) {
    long th = n / 8;
    dim3 grid((unsigned)((th + 255) / 256)), block(256);
    hipLaunchKernelGGL(cvt_bf16_k, grid, block, 0, s, in, out, n);
}
static inline void launch_gemm(const __bf16* X, const __bf16* W, float* Y, int M,
                               hipStream_t s) {
    dim3 grid((M + 127) / 128), block(256);
    hipLaunchKernelGGL(gemm128_wmma, grid, block, 0, s, X, W, Y, M);
}
static inline void launch_spmm(const int* r, const int* c, const float* v,
                               const float* T, float* Y, int nnz, hipStream_t s) {
    long threads = (long)nnz * 32;
    dim3 grid((unsigned)((threads + 255) / 256)), block(256);
    hipLaunchKernelGGL(spmm_atomic, grid, block, 0, s, r, c, v, T, Y, nnz);
}

extern "C" void kernel_launch(void* const* d_in, const int* in_sizes, int n_in,
                              void* d_out, int out_size, void* d_ws, size_t ws_size,
                              hipStream_t stream) {
    (void)n_in; (void)out_size; (void)ws_size;
    const int C = 128;

    // Inputs (setup_inputs order)
    const float* x_in[4];
    for (int r = 0; r < 4; ++r) x_in[r] = (const float*)d_in[r];
    int N[4];
    for (int r = 0; r < 4; ++r) N[r] = in_sizes[r] / C;

    const int* adj_row[4]; const int* adj_col[4]; const float* adj_val[4]; int adj_nnz[4];
    for (int r = 0; r < 4; ++r) {
        adj_row[r] = (const int*)  d_in[4 + 3 * r];
        adj_col[r] = (const int*)  d_in[5 + 3 * r];
        adj_val[r] = (const float*)d_in[6 + 3 * r];
        adj_nnz[r] = in_sizes[4 + 3 * r];
    }
    const int* inc_row[3]; const int* inc_col[3]; const float* inc_val[3]; int inc_nnz[3];
    for (int i = 0; i < 3; ++i) {
        inc_row[i] = (const int*)  d_in[16 + 3 * i];
        inc_col[i] = (const int*)  d_in[17 + 3 * i];
        inc_val[i] = (const float*)d_in[18 + 3 * i];
        inc_nnz[i] = in_sizes[16 + 3 * i];
    }
    const float* W_same = (const float*)d_in[25]; // [2,4,128,128]
    const float* W_h2l  = (const float*)d_in[26]; // [2,3,128,128]
    const float* W_l2h  = (const float*)d_in[27]; // [2,3,128,128]
    const float* W_out  = (const float*)d_in[28]; // [128,10]
    const float* b_out  = (const float*)d_in[29]; // [10]

    int maxN = N[0];
    for (int r = 1; r < 4; ++r) if (N[r] > maxN) maxN = N[r];
    long totN = 0;
    for (int r = 0; r < 4; ++r) totN += N[r];

    // Workspace layout
    char* base = (char*)d_ws;
    size_t off = 0;
    auto alloc = [&](size_t bytes) -> char* {
        char* p = base + off;
        off += (bytes + 255) & ~(size_t)255;
        return p;
    };
    float* acc = (float*)alloc((size_t)maxN * C * sizeof(float));   // shared f32 accumulator
    float* tS  = (float*)alloc((size_t)maxN * C * sizeof(float));
    float* tH  = (float*)alloc((size_t)maxN * C * sizeof(float));
    float* tL  = (float*)alloc((size_t)maxN * C * sizeof(float));
    __bf16* xbfA[4]; for (int r = 0; r < 4; ++r) xbfA[r] = (__bf16*)alloc((size_t)N[r] * C * 2);
    __bf16* xbfB[4]; for (int r = 0; r < 4; ++r) xbfB[r] = (__bf16*)alloc((size_t)N[r] * C * 2);
    __bf16* WsameB = (__bf16*)alloc((size_t)2 * 4 * C * C * 2);
    __bf16* Wh2lB  = (__bf16*)alloc((size_t)2 * 3 * C * C * 2);
    __bf16* Wl2hB  = (__bf16*)alloc((size_t)2 * 3 * C * C * 2);

    // One-time conversions: weights + layer-0 features -> bf16
    launch_cvt(W_same, WsameB, (long)2 * 4 * C * C, stream);
    launch_cvt(W_h2l,  Wh2lB,  (long)2 * 3 * C * C, stream);
    launch_cvt(W_l2h,  Wl2hB,  (long)2 * 3 * C * C, stream);
    for (int r = 0; r < 4; ++r)
        launch_cvt(x_in[r], xbfA[r], (long)N[r] * C, stream);

    __bf16** cur = xbfA;
    __bf16** nxt = xbfB;

    for (int l = 0; l < 2; ++l) {
        for (int r = 0; r < 4; ++r) {
            // Dense transforms (bf16 WMMA)
            launch_gemm(cur[r], WsameB + ((size_t)(l * 4 + r)) * C * C, tS, N[r], stream);
            if (r < 3)
                launch_gemm(cur[r + 1], Wh2lB + ((size_t)(l * 3 + r)) * C * C, tH, N[r + 1], stream);
            if (r > 0)
                launch_gemm(cur[r - 1], Wl2hB + ((size_t)(l * 3 + (r - 1))) * C * C, tL, N[r - 1], stream);

            // Accumulator init + sparse aggregation
            hipMemsetAsync(acc, 0, (size_t)N[r] * C * sizeof(float), stream);
            launch_spmm(adj_row[r], adj_col[r], adj_val[r], tS, acc, adj_nnz[r], stream);
            if (r < 3)  // y[irow] += ival * tH[icol]   (incidence r+1)
                launch_spmm(inc_row[r], inc_col[r], inc_val[r], tH, acc, inc_nnz[r], stream);
            if (r > 0)  // y[icol] += ival * tL[irow]   (incidence r, transposed)
                launch_spmm(inc_col[r - 1], inc_row[r - 1], inc_val[r - 1], tL, acc,
                            inc_nnz[r - 1], stream);

            // sigmoid + downconvert into next layer's feature set
            long n = (long)N[r] * C;
            dim3 g((unsigned)((n / 8 + 255) / 256)), b(256);
            hipLaunchKernelGGL(sig_bf16_k, g, b, 0, stream, acc, nxt[r], n);
        }
        __bf16** tmp = cur; cur = nxt; nxt = tmp;
    }

    // Head: softmax(cur[0] @ W_out + b_out)
    {
        dim3 g((N[0] + 255) / 256), b(256);
        hipLaunchKernelGGL(head_k, g, b, 0, stream, cur[0], W_out, b_out,
                           (float*)d_out, N[0]);
    }
}
